// Decoder_79044578115606
// MI455X (gfx1250) — matrix operand order repaired and checked by reading, not compiled, not run on hardware
//
#include <hip/hip_runtime.h>

#define LSEQ 64
#define BSZ  128
#define HID  256
#define VOC  32000
#define G4   1024
#define GO_IDX (VOC - 1)

typedef float  v2f   __attribute__((ext_vector_type(2)));
typedef float  v8f   __attribute__((ext_vector_type(8)));
typedef __bf16 v16bf __attribute__((ext_vector_type(16)));

union BfFrag { v16bf v; uint4 q[2]; };

__device__ __forceinline__ float sigm(float x) { return 1.0f / (1.0f + __expf(-x)); }

__device__ __forceinline__ unsigned short f32_to_bf16(float f) {
    unsigned int u = __builtin_bit_cast(unsigned int, f);
    unsigned int r = u + 0x7FFFu + ((u >> 16) & 1u);   // round-to-nearest-even
    return (unsigned short)(r >> 16);
}

// ---------------------------------------------------------------- embedding
__global__ void k_embed(const int* __restrict__ outputs,
                        const float* __restrict__ embed_W,
                        float* __restrict__ X) {
    int idx  = blockIdx.x * blockDim.x + threadIdx.x;   // over L*B*H
    int hcol = idx % HID;
    int row  = idx / HID;                               // t*B + b
    int t    = row / BSZ;
    int tok  = (t == 0) ? GO_IDX : outputs[row];
    X[idx] = embed_W[(size_t)tok * HID + hcol];
}

// ---------------------------------------------------------------- state init
__global__ void k_init(const float* __restrict__ h0, const float* __restrict__ c0,
                       float* __restrict__ h, float* __restrict__ c) {
    int i = blockIdx.x * blockDim.x + threadIdx.x;      // over B*H
    h[i] = h0[i];
    c[i] = c0[i];
}

// ---------------------------------------------------------------- f32 -> bf16
__global__ void k_cvt_bf16(const float* __restrict__ src,
                           unsigned short* __restrict__ dst) {
    int i = blockIdx.x * blockDim.x + threadIdx.x;
    dst[i] = f32_to_bf16(src[i]);
}

// ---------------------------------------------------------------- Xg = X @ W_ih^T + (b_ih + b_hh)
// fp32 WMMA 16x16x4.  One wave -> 16x64 tile (4 accumulators).
__global__ void k_xg(const float* __restrict__ X, const float* __restrict__ W_ih,
                     const float* __restrict__ b_ih, const float* __restrict__ b_hh,
                     float* __restrict__ Xg) {
    const int lane  = threadIdx.x & 31;
    const int wave  = blockIdx.x * (blockDim.x >> 5) + (threadIdx.x >> 5);
    const int mtile = wave >> 4;            // 512 tiles of 16 rows
    const int ngrp  = wave & 15;            // 16 groups of 64 cols
    const int m0 = mtile * 16, n0 = ngrp * 64;
    const int lm = lane & 15, g = lane >> 4;
    const int m  = m0 + lm;

    v8f acc[4] = {};
    for (int k0 = 0; k0 < HID; k0 += 4) {
        const int kb = k0 + 2 * g;
        v2f a = *(const v2f*)(X + (size_t)m * HID + kb);
#pragma unroll
        for (int j = 0; j < 4; ++j) {
            v2f b = *(const v2f*)(W_ih + (size_t)(n0 + 16 * j + lm) * HID + kb);
            acc[j] = __builtin_amdgcn_wmma_f32_16x16x4_f32(
                false, a, false, b, (short)0, acc[j], false, false);
        }
    }
#pragma unroll
    for (int j = 0; j < 4; ++j) {
        const int n    = n0 + 16 * j + lm;
        const float bv = b_ih[n] + b_hh[n];
#pragma unroll
        for (int r = 0; r < 8; ++r) {
            const int mr = m0 + r + 8 * g;
            Xg[(size_t)mr * G4 + n] = acc[j][r] + bv;
        }
    }
}

// ---------------------------------------------------------------- one LSTM step
// gates = h_in @ W_hh^T + Xg[t]; cell update in registers; h stored f32 + bf16.
__global__ void k_step(int t, const float* __restrict__ hin, float* __restrict__ hout,
                       float* __restrict__ c, const float* __restrict__ W_hh,
                       const float* __restrict__ Xg,
                       unsigned short* __restrict__ predbf) {
    const int lane  = threadIdx.x & 31;
    const int wave  = blockIdx.x * (blockDim.x >> 5) + (threadIdx.x >> 5);
    const int mtile = wave >> 4;            // 8 tiles over B=128
    const int ntile = wave & 15;            // 16 tiles over H=256
    const int m0 = mtile * 16, n0 = ntile * 16;
    const int lm = lane & 15, g = lane >> 4;
    const int m  = m0 + lm;
    const int n  = n0 + lm;

    v8f acc[4] = {};                        // i, f, g, o accumulators
    for (int k0 = 0; k0 < HID; k0 += 4) {
        const int kb = k0 + 2 * g;
        v2f a = *(const v2f*)(hin + (size_t)m * HID + kb);
#pragma unroll
        for (int gi = 0; gi < 4; ++gi) {
            v2f b = *(const v2f*)(W_hh + (size_t)(gi * HID + n) * HID + kb);
            acc[gi] = __builtin_amdgcn_wmma_f32_16x16x4_f32(
                false, a, false, b, (short)0, acc[gi], false, false);
        }
    }
#pragma unroll
    for (int r = 0; r < 8; ++r) {
        const int mr = m0 + r + 8 * g;
        const size_t grow = (size_t)(t * BSZ + mr) * G4 + n;
        float iv = acc[0][r] + Xg[grow];
        float fv = acc[1][r] + Xg[grow + 256];
        float gv = acc[2][r] + Xg[grow + 512];
        float ov = acc[3][r] + Xg[grow + 768];
        const size_t sidx = (size_t)mr * HID + n;
        float cn = sigm(fv) * c[sidx] + sigm(iv) * tanhf(gv);
        float hn = sigm(ov) * tanhf(cn);
        c[sidx]    = cn;
        hout[sidx] = hn;
        predbf[(size_t)(t * BSZ + mr) * HID + n] = f32_to_bf16(hn);
    }
}

// ---------------------------------------------------------------- logits = pred @ proj_W^T + proj_b
// bf16 WMMA 16x16x32, f32 accumulate.  One wave -> 16x64 tile; NT stores (1 GB stream).
__global__ void k_proj(const unsigned short* __restrict__ predbf,
                       const unsigned short* __restrict__ wbf,
                       const float* __restrict__ proj_b,
                       float* __restrict__ out) {
    const int lane  = threadIdx.x & 31;
    const int wave  = blockIdx.x * (blockDim.x >> 5) + (threadIdx.x >> 5);
    const int mtile = wave / 500;           // 512 tiles of 16 rows
    const int ngrp  = wave % 500;           // 500 groups of 64 cols (500*64 = 32000)
    const int m0 = mtile * 16, n0 = ngrp * 64;
    const int lm = lane & 15, g = lane >> 4;

    const unsigned short* arow = predbf + (size_t)(m0 + lm) * HID;
    v8f acc[4] = {};
    for (int k0 = 0; k0 < HID; k0 += 32) {
        BfFrag A;                            // lane holds K = [k0+8g, +8) and [k0+16+8g, +8)
        A.q[0] = *(const uint4*)(arow + k0 + 8 * g);
        A.q[1] = *(const uint4*)(arow + k0 + 16 + 8 * g);
#pragma unroll
        for (int j = 0; j < 4; ++j) {
            const unsigned short* brow =
                wbf + (size_t)(n0 + 16 * j + lm) * HID + k0 + 8 * g;  // K = [k0+8g, +8) and [k0+16+8g, +8)
            BfFrag Bf;
            Bf.q[0] = *(const uint4*)(brow);
            Bf.q[1] = *(const uint4*)(brow + 16);
            acc[j] = __builtin_amdgcn_wmma_f32_16x16x32_bf16(
                false, A.v, false, Bf.v, (short)0, acc[j], false, false);
        }
    }
#pragma unroll
    for (int j = 0; j < 4; ++j) {
        const int n    = n0 + 16 * j + lm;
        const float bv = proj_b[n];
#pragma unroll
        for (int r = 0; r < 8; ++r) {
            const int mr = m0 + r + 8 * g;
            __builtin_nontemporal_store(acc[j][r] + bv, out + (size_t)mr * VOC + n);
        }
    }
}

// ================================================================ host
extern "C" void kernel_launch(void* const* d_in, const int* in_sizes, int n_in,
                              void* d_out, int out_size, void* d_ws, size_t ws_size,
                              hipStream_t stream) {
    const int*   outputs = (const int*)  d_in[0];
    const float* h0      = (const float*)d_in[1];
    const float* c0      = (const float*)d_in[2];
    /* d_in[3] = batch_size scalar (compile-time constant here) */
    const float* embed_W = (const float*)d_in[4];
    const float* W_ih    = (const float*)d_in[5];
    const float* W_hh    = (const float*)d_in[6];
    const float* b_ih    = (const float*)d_in[7];
    const float* b_hh    = (const float*)d_in[8];
    const float* proj_W  = (const float*)d_in[9];
    const float* proj_b  = (const float*)d_in[10];
    float* out = (float*)d_out;

    char* ws = (char*)d_ws;
    size_t o = 0;
    float* X   = (float*)(ws + o);           o += (size_t)LSEQ * BSZ * HID * 4;   // 8 MB
    float* Xg  = (float*)(ws + o);           o += (size_t)LSEQ * BSZ * G4  * 4;   // 32 MB
    float* hA  = (float*)(ws + o);           o += (size_t)BSZ * HID * 4;
    float* hB  = (float*)(ws + o);           o += (size_t)BSZ * HID * 4;
    float* cS  = (float*)(ws + o);           o += (size_t)BSZ * HID * 4;
    unsigned short* predbf = (unsigned short*)(ws + o); o += (size_t)LSEQ * BSZ * HID * 2; // 4 MB
    unsigned short* wbf    = (unsigned short*)(ws + o); o += (size_t)VOC * HID * 2;        // 16 MB

    k_init <<<(BSZ * HID) / 256, 256, 0, stream>>>(h0, c0, hA, cS);
    k_embed<<<(LSEQ * BSZ * HID) / 256, 256, 0, stream>>>(outputs, embed_W, X);
    k_cvt_bf16<<<(VOC * HID) / 256, 256, 0, stream>>>(proj_W, wbf);

    // 512 mtiles * 16 ngroups = 8192 waves, 8 waves/block
    k_xg<<<1024, 256, 0, stream>>>(X, W_ih, b_ih, b_hh, Xg);

    // sequential recurrence, h ping-pong
    for (int t = 0; t < LSEQ; ++t) {
        const float* hin = (t & 1) ? hB : hA;
        float*      hout = (t & 1) ? hA : hB;
        k_step<<<16, 256, 0, stream>>>(t, hin, hout, cS, W_hh, Xg, predbf);
    }

    // 512 mtiles * 500 ngroups = 256000 waves, 8 waves/block
    k_proj<<<32000, 256, 0, stream>>>(predbf, wbf, proj_b, out);
}
